// MoE_78168404787630
// MI455X (gfx1250) — matrix operand order repaired and checked by reading, not compiled, not run on hardware
//
#include <hip/hip_runtime.h>

// ---------------- problem constants ----------------
#define TTOK   32768      // B*S = 8*4096
#define DIMK   1024       // model dim
#define INTERN 512        // routed expert inter dim
#define SHIN   1024       // shared expert inter dim (= 2 pseudo experts of 512)
#define NEXP   32
#define NPSE   34         // 32 routed + 2 shared pseudo-experts
#define TOPK   4
#define NGROUP 8
#define TOPKG  4
// max tiles: routed padded (8192+32) + shared pseudo (2*2048)
#define MAXTILES (TTOK*TOPK/16 + NEXP + 2*(TTOK/16))
#define LTOK_N   (TTOK*TOPK + NEXP*16 + 2*TTOK)

typedef __attribute__((ext_vector_type(16))) __bf16 v16bf;
typedef __attribute__((ext_vector_type(8)))  __bf16 bf16x8;
typedef __attribute__((ext_vector_type(8)))  float  v8f;

__device__ __forceinline__ v16bf mkfrag(bf16x8 lo, bf16x8 hi) {
  v16bf r;
#pragma unroll
  for (int i = 0; i < 8; ++i) { r[i] = lo[i]; r[i + 8] = hi[i]; }
  return r;
}

__device__ __forceinline__ v16bf ldfragB(const __bf16* p, unsigned fo) {
  // one WMMA B fragment = 32 contiguous bytes per lane in the pre-laid layout
  return mkfrag(*(const bf16x8*)(p + fo), *(const bf16x8*)(p + fo + 8));
}

__device__ __forceinline__ v8f wmma_bf16(v16bf a, v16bf b, v8f c) {
  // D(f32 16x16) = A(16x32 bf16) * B(32x16 bf16) + C
  return __builtin_amdgcn_wmma_f32_16x16x32_bf16(false, a, false, b, (short)0, c,
                                                 false, false);
}

// ---------------- prep kernels ----------------
__global__ void k_zero(int* ctrl) {
  if (threadIdx.x < 64) ctrl[threadIdx.x] = 0;  // cnt[32] + cur[32]
}

__global__ void k_cvt(const float* __restrict__ s, __bf16* __restrict__ d, size_t n) {
  size_t i = (size_t)blockIdx.x * 256 + threadIdx.x;
  if (i < n) d[i] = (__bf16)s[i];
}

// src: nmat row-major [K][N] f32  ->  dst: WMMA B-fragment layout
// [nmat][N/16][K/32][lane(32)][half(16)] bf16, where lane l, half h holds
// B[kc*32 + (l>=16?16:0) + h][j*16 + (l&15)]   (contiguous k order, same as the A reads)
__global__ void k_layout(const float* __restrict__ src, __bf16* __restrict__ dst,
                         int K, int N, int nmat) {
  size_t idx = (size_t)blockIdx.x * 256 + threadIdx.x;
  size_t per = (size_t)K * N;
  size_t tot = per * nmat;
  if (idx >= tot) return;
  size_t mat = idx / per;
  size_t r = idx % per;
  int h    = (int)(r & 15);
  int l    = (int)((r >> 4) & 31);
  int frag = (int)(r >> 9);
  int kt   = K >> 5;
  int kc   = frag % kt;
  int j    = frag / kt;
  int k    = kc * 32 + ((l >> 4) << 4) + h;
  int n    = j * 16 + (l & 15);
  dst[idx] = (__bf16)src[mat * per + (size_t)k * N + n];
}

// ---------------- gate (exact fp32 routing) ----------------
__global__ __launch_bounds__(256) void k_gate(const float* __restrict__ x,
                                              const float* __restrict__ gw,
                                              const float* __restrict__ gb,
                                              int* __restrict__ oidx,
                                              float* __restrict__ owt) {
  __shared__ float probs[8][32];
  int wv = threadIdx.x >> 5, lane = threadIdx.x & 31;
  int t = blockIdx.x * 8 + wv;
  const float* xr = x + (size_t)t * DIMK;
  float s = gb[lane];
#pragma unroll 4
  for (int k = 0; k < DIMK; ++k) s = fmaf(xr[k], gw[k * NEXP + lane], s);
  // softmax across the 32 experts (one per lane, wave32)
  float m = s;
#pragma unroll
  for (int o = 16; o; o >>= 1) m = fmaxf(m, __shfl_xor(m, o, 32));
  float p = __expf(s - m);
  float sum = p;
#pragma unroll
  for (int o = 16; o; o >>= 1) sum += __shfl_xor(sum, o, 32);
  probs[wv][lane] = p / sum;
  __syncthreads();
  if (lane == 0) {
    float* pr = probs[wv];
    float gs[NGROUP];
    for (int g = 0; g < NGROUP; ++g) {  // sum of top-2 of the 4 experts in group
      float v0 = pr[4 * g], v1 = pr[4 * g + 1], v2 = pr[4 * g + 2], v3 = pr[4 * g + 3];
      float a = fmaxf(v0, v1), b = fminf(v0, v1);
      float c = fmaxf(v2, v3), d = fminf(v2, v3);
      float m1 = fmaxf(a, c);
      float m2 = (a > c) ? fmaxf(b, c) : fmaxf(d, a);
      gs[g] = m1 + m2;
    }
    bool gsel[NGROUP] = {};
    for (int it = 0; it < TOPKG; ++it) {  // strict > ascending = lowest-index ties
      int bi = 0; float bv = -1e30f;
      for (int g = 0; g < NGROUP; ++g)
        if (!gsel[g] && gs[g] > bv) { bv = gs[g]; bi = g; }
      gsel[bi] = true;
    }
    bool taken[NEXP] = {};
    for (int sl = 0; sl < TOPK; ++sl) {
      int bi = 0; float bv = -1e30f;
      for (int e = 0; e < NEXP; ++e)
        if (gsel[e >> 2] && !taken[e] && pr[e] > bv) { bv = pr[e]; bi = e; }
      taken[bi] = true;
      oidx[t * TOPK + sl] = bi;
      owt[t * TOPK + sl] = pr[bi];  // ROUTE_SCALE = 1.0
    }
  }
}

// ---------------- dispatch ----------------
__global__ void k_count(const int* __restrict__ idx, int* __restrict__ cnt) {
  int t = blockIdx.x * 256 + threadIdx.x;
  if (t >= TTOK) return;
#pragma unroll
  for (int s = 0; s < TOPK; ++s) atomicAdd(&cnt[idx[t * TOPK + s]], 1);
}

__global__ void k_offsets(const int* __restrict__ cnt, int* __restrict__ offs,
                          int* __restrict__ ltok, int* __restrict__ te,
                          int* __restrict__ tb, int* __restrict__ ntl) {
  if (threadIdx.x | blockIdx.x) return;
  int off = 0, tl = 0;
  for (int e = 0; e < NEXP; ++e) {
    offs[e] = off;
    int c = cnt[e];
    int pd = (c + 15) & ~15;
    for (int i = c; i < pd; ++i) ltok[off + i] = -1;  // padding rows
    for (int t = 0; t < pd; t += 16) { te[tl] = e; tb[tl] = off + t; ++tl; }
    off += pd;
  }
  for (int e = NEXP; e < NPSE; ++e) {  // shared pseudo-experts: all tokens
    offs[e] = off;
    for (int t = 0; t < TTOK; t += 16) { te[tl] = e; tb[tl] = off + t; ++tl; }
    off += TTOK;
  }
  ntl[0] = tl;
}

__global__ void k_scatter(const int* __restrict__ idx, const float* __restrict__ wts,
                          const int* __restrict__ offs, int* __restrict__ cur,
                          int* __restrict__ ltok, float* __restrict__ lwt) {
  int t = blockIdx.x * 256 + threadIdx.x;
  if (t >= TTOK) return;
#pragma unroll
  for (int s = 0; s < TOPK; ++s) {
    int e = idx[t * TOPK + s];
    int pos = atomicAdd(&cur[e], 1);
    int o = offs[e] + pos;
    ltok[o] = t;
    lwt[o] = wts[t * TOPK + s];
  }
}

__global__ void k_fill(const int* __restrict__ offs, int* __restrict__ ltok,
                       float* __restrict__ lwt) {
  int i = blockIdx.x * 256 + threadIdx.x;
  if (i >= 2 * TTOK) return;
  int h = i / TTOK, t = i % TTOK;
  int o = offs[NEXP + h] + t;
  ltok[o] = t;
  lwt[o] = 1.0f;
}

// ---------------- fused SwiGLU expert FFN (bf16 WMMA) ----------------
// One block = 16-token tile of one (pseudo-)expert. 8 waves:
// stage1: g,u = x@w1/w3 (K=1024 -> 32 WMMA steps, 4 n-tiles/wave), SiLU -> LDS h
// stage2: out = h@w2 (K=512, 8 n-tiles/wave), weighted atomic scatter into y.
__global__ __launch_bounds__(256) void k_ffn(
    const __bf16* __restrict__ xb,
    const __bf16* __restrict__ w1b, const __bf16* __restrict__ w3b,
    const __bf16* __restrict__ w2b,
    const __bf16* __restrict__ sw1b, const __bf16* __restrict__ sw3b,
    const __bf16* __restrict__ sw2b,
    const float* __restrict__ b1, const float* __restrict__ b3,
    const float* __restrict__ b2,
    const float* __restrict__ sb1, const float* __restrict__ sb3,
    const float* __restrict__ sb2,
    const int* __restrict__ te, const int* __restrict__ tb,
    const int* __restrict__ ntl,
    const int* __restrict__ ltok, const float* __restrict__ lwt,
    float* __restrict__ y) {
  constexpr int XSTR = DIMK + 8;    // halfs; row stride 2064B (16B aligned, bank-skewed)
  constexpr int HSTR = INTERN + 8;  // halfs; row stride 1040B
  __shared__ __align__(16) __bf16 xt[16 * XSTR];
  __shared__ __align__(16) __bf16 hl[16 * HSTR];
  __shared__ int stok[16];
  __shared__ float swt[16];

  int bid = blockIdx.x;
  if (bid >= ntl[0]) return;  // uniform exit
  int e = te[bid];
  int base = tb[bid];
  int tid = threadIdx.x;
  if (tid < 16) { stok[tid] = ltok[base + tid]; swt[tid] = lwt[base + tid]; }
  __syncthreads();

  // stage the 16x1024 bf16 activation tile into LDS (coalesced 16B chunks)
#pragma unroll
  for (int rep = 0; rep < 8; ++rep) {
    int c = rep * 256 + tid;         // 2048 16B chunks
    int row = c >> 7;                // 128 chunks per row
    int col = (c & 127) << 3;        // half offset
    int tk = stok[row];
    bf16x8 v;
    if (tk >= 0) v = *(const bf16x8*)(xb + (size_t)tk * DIMK + col);
    else {
#pragma unroll
      for (int i = 0; i < 8; ++i) v[i] = (__bf16)0.0f;
    }
    *(bf16x8*)(xt + row * XSTR + col) = v;
  }
  __syncthreads();

  int wv = tid >> 5, lane = tid & 31, mrow = lane & 15, hi = lane >> 4;

  const __bf16 *W1, *W3, *W2;
  const float *B1, *B3, *B2;
  int KT2, KB2;
  if (e < NEXP) {
    W1 = w1b + (size_t)e * DIMK * INTERN;
    W3 = w3b + (size_t)e * DIMK * INTERN;
    W2 = w2b + (size_t)e * INTERN * DIMK;
    B1 = b1 + e * INTERN; B3 = b3 + e * INTERN; B2 = b2 + e * DIMK;
    KT2 = INTERN / 32; KB2 = 0;
  } else {
    int h = e - NEXP;  // shared pseudo-expert half
    W1 = sw1b + (size_t)h * DIMK * INTERN;  // j-major layout => halves contiguous
    W3 = sw3b + (size_t)h * DIMK * INTERN;
    W2 = sw2b;
    B1 = sb1 + h * INTERN; B3 = sb3 + h * INTERN;
    B2 = (h == 0) ? sb2 : nullptr;  // add sb2 exactly once
    KT2 = SHIN / 32; KB2 = h * (INTERN / 32);
  }

  // ---- stage 1: g = x@w1+b1, u = x@w3+b3, h = silu(g)*u ----
  {
    v8f ag[4], au[4];
#pragma unroll
    for (int nt = 0; nt < 4; ++nt)
#pragma unroll
      for (int i = 0; i < 8; ++i) { ag[nt][i] = 0.f; au[nt][i] = 0.f; }
    int jbase = wv * 4;  // 32 n-tiles over 8 waves
    // per-n-tile 32-bit base offsets: +kc*512 folds into the load immediates
    unsigned fo0[4];
#pragma unroll
    for (int nt = 0; nt < 4; ++nt)
      fo0[nt] = (unsigned)((jbase + nt) * (DIMK / 32)) * 512u + (unsigned)lane * 16u;
#pragma unroll 4
    for (int kc = 0; kc < DIMK / 32; ++kc) {
      const __bf16* ap = xt + mrow * XSTR + kc * 32 + hi * 16;
      v16bf a = mkfrag(*(const bf16x8*)ap, *(const bf16x8*)(ap + 8));
      v16bf bw[8];  // distinct regs: all 16 b128 loads issue before the WMMAs
#pragma unroll
      for (int nt = 0; nt < 4; ++nt) {
        unsigned fo = fo0[nt] + (unsigned)kc * 512u;
        bw[nt]     = ldfragB(W1, fo);
        bw[4 + nt] = ldfragB(W3, fo);
      }
#pragma unroll
      for (int nt = 0; nt < 4; ++nt) {
        ag[nt] = wmma_bf16(a, bw[nt], ag[nt]);
        au[nt] = wmma_bf16(a, bw[4 + nt], au[nt]);
      }
    }
#pragma unroll
    for (int nt = 0; nt < 4; ++nt) {
      int n = (jbase + nt) * 16 + mrow;  // this lane's output column
      float bb1 = B1[n], bb3 = B3[n];
#pragma unroll
      for (int r = 0; r < 8; ++r) {  // C row m = r + 8*hi
        float g = ag[nt][r] + bb1;
        float u = au[nt][r] + bb3;
        float hv = (g / (1.0f + __expf(-g))) * u;  // silu(g)*u
        hl[(r + 8 * hi) * HSTR + n] = (__bf16)hv;
      }
    }
  }
  __syncthreads();

  // ---- stage 2: out = h@w2 (+b2), weighted atomic combine ----
#pragma unroll 1
  for (int ch = 0; ch < 2; ++ch) {
    v8f ao[4];
#pragma unroll
    for (int nt = 0; nt < 4; ++nt)
#pragma unroll
      for (int i = 0; i < 8; ++i) ao[nt][i] = 0.f;
    int jbase = ch * 32 + wv * 4;  // 64 n-tiles over 8 waves, 2 chunks
    unsigned fo0[4];
#pragma unroll
    for (int nt = 0; nt < 4; ++nt)
      fo0[nt] = (unsigned)((jbase + nt) * KT2 + KB2) * 512u + (unsigned)lane * 16u;
#pragma unroll 4
    for (int kc = 0; kc < INTERN / 32; ++kc) {
      const __bf16* ap = hl + mrow * HSTR + kc * 32 + hi * 16;
      v16bf a = mkfrag(*(const bf16x8*)ap, *(const bf16x8*)(ap + 8));
      v16bf bw[4];
#pragma unroll
      for (int nt = 0; nt < 4; ++nt)
        bw[nt] = ldfragB(W2, fo0[nt] + (unsigned)kc * 512u);
#pragma unroll
      for (int nt = 0; nt < 4; ++nt) ao[nt] = wmma_bf16(a, bw[nt], ao[nt]);
    }
#pragma unroll
    for (int nt = 0; nt < 4; ++nt) {
      int n = (jbase + nt) * 16 + mrow;
      float bb2 = B2 ? B2[n] : 0.0f;
#pragma unroll
      for (int r = 0; r < 8; ++r) {
        int row = r + 8 * hi;
        int tk = stok[row];
        if (tk >= 0) {
          float val = swt[row] * (ao[nt][r] + bb2);
          __hip_atomic_fetch_add(&y[(size_t)tk * DIMK + n], val,
                                 __ATOMIC_RELAXED, __HIP_MEMORY_SCOPE_AGENT);
        }
      }
    }
  }
}

// ---------------- host launcher ----------------
extern "C" void kernel_launch(void* const* d_in, const int* in_sizes, int n_in,
                              void* d_out, int out_size, void* d_ws, size_t ws_size,
                              hipStream_t stream) {
  (void)in_sizes; (void)n_in; (void)ws_size;
  const float* x   = (const float*)d_in[0];
  const float* gw  = (const float*)d_in[1];
  const float* gb  = (const float*)d_in[2];
  const float* w1  = (const float*)d_in[3];
  const float* b1  = (const float*)d_in[4];
  const float* w3  = (const float*)d_in[5];
  const float* b3  = (const float*)d_in[6];
  const float* w2  = (const float*)d_in[7];
  const float* b2  = (const float*)d_in[8];
  const float* sw1 = (const float*)d_in[9];
  const float* sb1 = (const float*)d_in[10];
  const float* sw3 = (const float*)d_in[11];
  const float* sb3 = (const float*)d_in[12];
  const float* sw2 = (const float*)d_in[13];
  const float* sb2 = (const float*)d_in[14];
  float* y = (float*)d_out;

  char* base = (char*)d_ws;
  size_t off = 0;
  auto carve = [&](size_t bytes) -> char* {
    char* r = base + off;
    off = (off + bytes + 255) & ~(size_t)255;
    return r;
  };
  __bf16* xb   = (__bf16*)carve((size_t)TTOK * DIMK * 2);
  __bf16* w1b  = (__bf16*)carve((size_t)NEXP * DIMK * INTERN * 2);
  __bf16* w3b  = (__bf16*)carve((size_t)NEXP * DIMK * INTERN * 2);
  __bf16* w2b  = (__bf16*)carve((size_t)NEXP * INTERN * DIMK * 2);
  __bf16* sw1b = (__bf16*)carve((size_t)DIMK * SHIN * 2);
  __bf16* sw3b = (__bf16*)carve((size_t)DIMK * SHIN * 2);
  __bf16* sw2b = (__bf16*)carve((size_t)SHIN * DIMK * 2);
  int*   idx   = (int*)carve((size_t)TTOK * TOPK * 4);
  float* wts   = (float*)carve((size_t)TTOK * TOPK * 4);
  int*   ctrl  = (int*)carve(1024);  // cnt[32] cur[32] offs[34] ntiles[1]
  int*   ltok  = (int*)carve((size_t)LTOK_N * 4);
  float* lwt   = (float*)carve((size_t)LTOK_N * 4);
  int*   te    = (int*)carve((size_t)MAXTILES * 4);
  int*   tbv   = (int*)carve((size_t)MAXTILES * 4);
  int* cnt = ctrl, *cur = ctrl + 32, *offs = ctrl + 64, *ntl = ctrl + 98;

  hipMemsetAsync(d_out, 0, (size_t)out_size * sizeof(float), stream);
  k_zero<<<1, 64, 0, stream>>>(ctrl);

  k_cvt<<<(TTOK * (size_t)DIMK) / 256, 256, 0, stream>>>(x, xb, (size_t)TTOK * DIMK);
  k_layout<<<((size_t)NEXP * DIMK * INTERN) / 256, 256, 0, stream>>>(w1, w1b, DIMK, INTERN, NEXP);
  k_layout<<<((size_t)NEXP * DIMK * INTERN) / 256, 256, 0, stream>>>(w3, w3b, DIMK, INTERN, NEXP);
  k_layout<<<((size_t)NEXP * INTERN * DIMK) / 256, 256, 0, stream>>>(w2, w2b, INTERN, DIMK, NEXP);
  k_layout<<<((size_t)DIMK * SHIN) / 256, 256, 0, stream>>>(sw1, sw1b, DIMK, SHIN, 1);
  k_layout<<<((size_t)DIMK * SHIN) / 256, 256, 0, stream>>>(sw3, sw3b, DIMK, SHIN, 1);
  k_layout<<<((size_t)SHIN * DIMK) / 256, 256, 0, stream>>>(sw2, sw2b, SHIN, DIMK, 1);

  k_gate<<<TTOK / 8, 256, 0, stream>>>(x, gw, gb, idx, wts);
  k_count<<<TTOK / 256, 256, 0, stream>>>(idx, cnt);
  k_offsets<<<1, 1, 0, stream>>>(cnt, offs, ltok, te, tbv, ntl);
  k_scatter<<<TTOK / 256, 256, 0, stream>>>(idx, wts, offs, cur, ltok, lwt);
  k_fill<<<(2 * TTOK) / 256, 256, 0, stream>>>(offs, ltok, lwt);

  k_ffn<<<MAXTILES, 256, 0, stream>>>(xb, w1b, w3b, w2b, sw1b, sw3b, sw2b,
                                      b1, b3, b2, sb1, sb3, sb2,
                                      te, tbv, ntl, ltok, lwt, y);
}
